// LightweightAttention_609885356530
// MI455X (gfx1250) — matrix operand order repaired and checked by reading, not compiled, not run on hardware
//
#include <hip/hip_runtime.h>
#include <hip/hip_bf16.h>

// ---------------------------------------------------------------------------
// Windowed "lightweight" attention, MI455X (gfx1250, wave32, WMMA bf16).
// Pipeline: qkv GEMM -> per-channel window attention -> proj GEMM.
// All matrix math through v_wmma_f32_16x16x32_bf16 (f32 accumulate).
// All intermediates in a single [sel][b][ch][n][l] bf16 layout so every
// producer writes 32B-contiguous half-wave stores (no write amplification).
// ---------------------------------------------------------------------------

typedef __attribute__((ext_vector_type(16))) __bf16 bf16x16;
typedef __attribute__((ext_vector_type(8)))  __bf16 bf16x8;
typedef __attribute__((ext_vector_type(8)))  float  f32x8;
typedef __attribute__((ext_vector_type(4)))  float  f32x4;

#define LDA 280   // LDS row stride (bf16) for 64x256 weight tiles (16B aligned, conflict-free)
#define LDB 40    // LDS row stride (bf16) for 64x32 activation tiles

static __device__ __forceinline__ f32x8 zero8() {
  f32x8 z = {0.f, 0.f, 0.f, 0.f, 0.f, 0.f, 0.f, 0.f};
  return z;
}
static __device__ __forceinline__ bf16x8 zerob8() {
  bf16x8 z = {(__bf16)0.f, (__bf16)0.f, (__bf16)0.f, (__bf16)0.f,
              (__bf16)0.f, (__bf16)0.f, (__bf16)0.f, (__bf16)0.f};
  return z;
}

static __device__ __forceinline__ f32x8 wmma_bf16(bf16x16 a, bf16x16 b, f32x8 c) {
  // D = A(16x32) * B(32x16) + C(16x16 f32)
  return __builtin_amdgcn_wmma_f32_16x16x32_bf16(false, a, false, b, (short)0, c, false, false);
}

// A-matrix fragment (16x32, 16-bit): lane holds row M=lane%16; half=lane/16.
// element e -> k = (e&7) + (e>>3)*16 + half*8  => two contiguous 16B loads.
static __device__ __forceinline__ bf16x16 frag_a(const __bf16* row, int kbase, int half) {
  bf16x8 lo = *(const bf16x8*)(row + kbase + half * 8);
  bf16x8 hi = *(const bf16x8*)(row + kbase + 16 + half * 8);
  return __builtin_shufflevector(lo, hi, 0,1,2,3,4,5,6,7,8,9,10,11,12,13,14,15);
}

// B-matrix fragment (32x16, 16-bit): lane holds col N=lane%16; half=lane/16.
// element e -> k = (e&7) + (e>>3)*16 + half*8  => two contiguous 16B loads.
// `row` points at B^T row (= column of B) stored k-minor.
static __device__ __forceinline__ bf16x16 frag_b(const __bf16* row, int kbase, int half) {
  bf16x8 lo = *(const bf16x8*)(row + kbase + half * 8);
  bf16x8 hi = *(const bf16x8*)(row + kbase + 16 + half * 8);
  return __builtin_shufflevector(lo, hi, 0,1,2,3,4,5,6,7,8,9,10,11,12,13,14,15);
}

static __device__ __forceinline__ bf16x8 cvt8(f32x4 a, f32x4 b) {
  bf16x8 r;
  r[0] = (__bf16)a[0]; r[1] = (__bf16)a[1]; r[2] = (__bf16)a[2]; r[3] = (__bf16)a[3];
  r[4] = (__bf16)b[0]; r[5] = (__bf16)b[1]; r[6] = (__bf16)b[2]; r[7] = (__bf16)b[3];
  return r;
}

// ---------------------------------------------------------------------------
// Kernel 1: QKV projection + windowing.
// Block = (o-tile 64, window n, batch b); 256 threads = 8 waves.
// Output: qkv[(((sel*2048) + b*256 + ch)*784 + n)*64 + l]  (bf16, l-minor)
// ---------------------------------------------------------------------------
__global__ __launch_bounds__(256) void qkv_win_kernel(
    const float* __restrict__ x, const float* __restrict__ w_qkv,
    const float* __restrict__ b_qkv, __bf16* __restrict__ qkv) {
  __shared__ __bf16 As[64 * LDA];
  __shared__ __bf16 Bs[64 * LDB];

  const int t = threadIdx.x;
  const int lane = t & 31, wv = t >> 5;
  const int half = lane >> 4, l16 = lane & 15;
  const int o0 = blockIdx.x * 64;     // 0..704 step 64 (768 rows)
  const int n  = blockIdx.y;          // window 0..783
  const int b  = blockIdx.z;
  const int i0 = (n / 28) * 8, j0 = (n % 28) * 8;

  { // stage full 64x256 weight tile as bf16
    const int r = t >> 2, cb = (t & 3) * 64;
    const float* wr = w_qkv + (size_t)(o0 + r) * 256 + cb;
    __bf16* dst = As + r * LDA + cb;
#pragma unroll
    for (int it = 0; it < 8; ++it) {
      f32x4 v0 = *(const f32x4*)(wr + it * 8);
      f32x4 v1 = *(const f32x4*)(wr + it * 8 + 4);
      *(bf16x8*)(dst + it * 8) = cvt8(v0, v1);
    }
  }

  const int msub = wv & 3, nsub = (wv >> 2) * 2;
  const int cl = t >> 3, rw = t & 7;  // B-tile loader role: channel, window row
  f32x8 acc0 = zero8(), acc1 = zero8();

  for (int kc = 0; kc < 8; ++kc) {
    const int c0 = kc * 32;
    __syncthreads();
    { // x tile: 32 channels x 64 pixels, transposed into Bs[l][c]
      const float* xr = x + (((size_t)b * 256 + c0 + cl) * 224 + (i0 + rw)) * 224 + j0;
      f32x4 v0 = *(const f32x4*)xr;
      f32x4 v1 = *(const f32x4*)(xr + 4);
      bf16x8 p = cvt8(v0, v1);
#pragma unroll
      for (int q = 0; q < 8; ++q) Bs[(rw * 8 + q) * LDB + cl] = p[q];
    }
    __syncthreads();
    bf16x16 a  = frag_a(As + (msub * 16 + l16) * LDA, c0, half);
    bf16x16 b0 = frag_b(Bs + (nsub * 16 + l16) * LDB, 0, half);
    bf16x16 b1 = frag_b(Bs + ((nsub + 1) * 16 + l16) * LDB, 0, half);
    acc0 = wmma_bf16(a, b0, acc0);
    acc1 = wmma_bf16(a, b1, acc1);
  }

  // epilogue: lanes sweep l (32B-contiguous per half-wave), r sweeps ch
  const int obase = o0 + msub * 16 + half * 8;
#pragma unroll
  for (int j = 0; j < 2; ++j) {
    f32x8 acc = j ? acc1 : acc0;
    const int l = (nsub + j) * 16 + l16;
#pragma unroll
    for (int r = 0; r < 8; ++r) {
      const int o   = obase + r;            // 0..767: sel = o>>8, ch = o&255
      const int sel = o >> 8, ch = o & 255;
      qkv[(((size_t)sel * 2048 + (size_t)b * 256 + ch) * 784 + n) * 64 + l] =
          (__bf16)(acc[r] + b_qkv[o]);
    }
  }
}

// ---------------------------------------------------------------------------
// Kernel 2: per-(b, ch) window attention.  S = Q^T K / sqrt(32)  (K-dim = n),
// softmax rows, O^T[l][n] = sum_m P[l][m] V[n][m].
// Q/K slabs staged+transposed through LDS (global reads are coalesced 128B
// rows); V fragments load straight from global ([n][l] is already k-minor).
// Block = (ch, b); 256 threads = 8 waves.
// ---------------------------------------------------------------------------
__global__ __launch_bounds__(256) void win_attn_kernel(
    const __bf16* __restrict__ qkv, __bf16* __restrict__ oout) {
  __shared__ __bf16 Qs[64 * LDB];
  __shared__ __bf16 Ks[64 * LDB];
  __shared__ float  Ssh[64 * 66];
  __shared__ __bf16 P[64 * 72];

  const int t = threadIdx.x, lane = t & 31, wv = t >> 5;
  const int half = lane >> 4, l16 = lane & 15;
  const int ch = blockIdx.x, b = blockIdx.y;

  const size_t chblk = 784ull * 64;
  const __bf16* qtile = qkv + ((size_t)b * 256 + ch) * chblk;                 // sel 0
  const __bf16* ktile = qkv + ((size_t)2048 + b * 256 + ch) * chblk;          // sel 1
  const __bf16* vtile = qkv + ((size_t)4096 + b * 256 + ch) * chblk;          // sel 2

  const int lt = wv & 3, mt = (wv >> 2) * 2;
  const int nl = t >> 3, seg = t & 7;          // loader role: n-within-slab, l-octet

  f32x8 s0 = zero8(), s1 = zero8();
  for (int ks = 0; ks < 25; ++ks) {            // n = 784 -> 24.5 slabs of 32
    const int ng = ks * 32 + nl;
    __syncthreads();
    {
      bf16x8 pq = zerob8(), pk = zerob8();
      if (ng < 784) {
        pq = *(const bf16x8*)(qtile + (size_t)ng * 64 + seg * 8);
        pk = *(const bf16x8*)(ktile + (size_t)ng * 64 + seg * 8);
      }
      if (ng + 64 < 784) {                     // warm GL2 two slabs ahead
        __builtin_prefetch(qtile + (size_t)(ng + 64) * 64 + seg * 8, 0, 1);
        __builtin_prefetch(ktile + (size_t)(ng + 64) * 64 + seg * 8, 0, 1);
      }
#pragma unroll
      for (int q = 0; q < 8; ++q) {            // transpose into [l][n_local]
        Qs[(seg * 8 + q) * LDB + nl] = pq[q];
        Ks[(seg * 8 + q) * LDB + nl] = pk[q];
      }
    }
    __syncthreads();
    bf16x16 a  = frag_a(Qs + (lt * 16 + l16) * LDB, 0, half);
    bf16x16 b0 = frag_b(Ks + (mt * 16 + l16) * LDB, 0, half);
    bf16x16 b1 = frag_b(Ks + ((mt + 1) * 16 + l16) * LDB, 0, half);
    s0 = wmma_bf16(a, b0, s0);
    s1 = wmma_bf16(a, b1, s1);
  }

  const float scale = 0.17677669529663689f;   // 1/sqrt(32)
#pragma unroll
  for (int r = 0; r < 8; ++r) {
    const int l = lt * 16 + half * 8 + r;
    Ssh[l * 66 + mt * 16 + l16]       = s0[r] * scale;
    Ssh[l * 66 + (mt + 1) * 16 + l16] = s1[r] * scale;
  }
  __syncthreads();

  if (t < 64) {                                // row-wise softmax -> P (bf16)
    float* row = Ssh + t * 66;
    float m = row[0];
    for (int i = 1; i < 64; ++i) m = fmaxf(m, row[i]);
    float s = 0.f;
    for (int i = 0; i < 64; ++i) { float e = __expf(row[i] - m); row[i] = e; s += e; }
    const float inv = 1.f / s;
    __bf16* prow = P + t * 72;
    for (int i = 0; i < 64; ++i) prow[i] = (__bf16)(row[i] * inv);
  }
  __syncthreads();

  bf16x16 afr[4][2];
#pragma unroll
  for (int lt2 = 0; lt2 < 4; ++lt2)
#pragma unroll
    for (int ks2 = 0; ks2 < 2; ++ks2)
      afr[lt2][ks2] = frag_a(P + (lt2 * 16 + l16) * 72, ks2 * 32, half);

  __bf16* obase = oout + ((size_t)b * 256 + ch) * chblk;

  for (int nt = wv; nt < 49; nt += 8) {        // 49*16 = 784 output rows (n)
    const int nrow = nt * 16 + l16;
    const __bf16* vrow = vtile + (size_t)nrow * 64;
    bf16x16 vb0 = frag_b(vrow, 0, half);
    bf16x16 vb1 = frag_b(vrow, 32, half);
    f32x8 oa[4] = {zero8(), zero8(), zero8(), zero8()};
#pragma unroll
    for (int lt2 = 0; lt2 < 4; ++lt2) {
      oa[lt2] = wmma_bf16(afr[lt2][0], vb0, oa[lt2]);
      oa[lt2] = wmma_bf16(afr[lt2][1], vb1, oa[lt2]);
    }
    __bf16* orow = obase + (size_t)nrow * 64;
#pragma unroll
    for (int lt2 = 0; lt2 < 4; ++lt2) {
      bf16x8 p;
#pragma unroll
      for (int r = 0; r < 8; ++r) p[r] = (__bf16)oa[lt2][r];
      *(bf16x8*)(orow + lt2 * 16 + half * 8) = p;   // 16B contiguous in l
    }
  }
}

// ---------------------------------------------------------------------------
// Kernel 3: output projection + window un-permute + bias -> d_out (f32).
// Block = (o-tile 64, window n, batch b).
// ---------------------------------------------------------------------------
__global__ __launch_bounds__(256) void proj_kernel(
    const __bf16* __restrict__ ain, const float* __restrict__ w_proj,
    const float* __restrict__ b_proj, float* __restrict__ out) {
  __shared__ __bf16 As[64 * LDA];
  __shared__ __bf16 Bs[64 * LDB];

  const int t = threadIdx.x, lane = t & 31, wv = t >> 5;
  const int half = lane >> 4, l16 = lane & 15;
  const int o0 = blockIdx.x * 64;     // 0..192 step 64
  const int n = blockIdx.y, b = blockIdx.z;

  { // stage 64x256 w_proj tile as bf16
    const int r = t >> 2, cb = (t & 3) * 64;
    const float* wr = w_proj + (size_t)(o0 + r) * 256 + cb;
    __bf16* dst = As + r * LDA + cb;
#pragma unroll
    for (int it = 0; it < 8; ++it) {
      f32x4 v0 = *(const f32x4*)(wr + it * 8);
      f32x4 v1 = *(const f32x4*)(wr + it * 8 + 4);
      *(bf16x8*)(dst + it * 8) = cvt8(v0, v1);
    }
  }

  const int msub = wv & 3, nsub = (wv >> 2) * 2;
  const int cl = t >> 3, rw = t & 7;
  f32x8 acc0 = zero8(), acc1 = zero8();

  for (int kc = 0; kc < 8; ++kc) {
    const int c0 = kc * 32;
    __syncthreads();
    { // attn-out tile: 32 channels x 64 l, transposed into Bs[l][c]
      const __bf16* ar = ain + (((size_t)b * 256 + c0 + cl) * 784 + n) * 64 + rw * 8;
      bf16x8 p = *(const bf16x8*)ar;
#pragma unroll
      for (int q = 0; q < 8; ++q) Bs[(rw * 8 + q) * LDB + cl] = p[q];
    }
    __syncthreads();
    bf16x16 a  = frag_a(As + (msub * 16 + l16) * LDA, c0, half);
    bf16x16 b0 = frag_b(Bs + (nsub * 16 + l16) * LDB, 0, half);
    bf16x16 b1 = frag_b(Bs + ((nsub + 1) * 16 + l16) * LDB, 0, half);
    acc0 = wmma_bf16(a, b0, acc0);
    acc1 = wmma_bf16(a, b1, acc1);
  }

  const int gi = n / 28, gj = n % 28;
  const int obch = o0 + msub * 16 + half * 8;
#pragma unroll
  for (int j = 0; j < 2; ++j) {
    f32x8 acc = j ? acc1 : acc0;
    const int l  = (nsub + j) * 16 + l16;
    const int ii = gi * 8 + (l >> 3), jj = gj * 8 + (l & 7);
#pragma unroll
    for (int r = 0; r < 8; ++r) {
      const int o = obch + r;
      out[(((size_t)b * 256 + o) * 224 + ii) * 224 + jj] = acc[r] + b_proj[o];
    }
  }
}

// ---------------------------------------------------------------------------
// Workspace map (bytes):
//   [0,           616,562,688)  Q,K,V : 3 * 8*256 * 784 * 64 * bf16
//   [616,562,688, 822,083,584)  attention output, layout [b][ch][n][l]
// ---------------------------------------------------------------------------
extern "C" void kernel_launch(void* const* d_in, const int* in_sizes, int n_in,
                              void* d_out, int out_size, void* d_ws, size_t ws_size,
                              hipStream_t stream) {
  (void)in_sizes; (void)n_in; (void)out_size; (void)ws_size;
  const float* x      = (const float*)d_in[0];
  const float* w_qkv  = (const float*)d_in[1];
  const float* b_qkv  = (const float*)d_in[2];
  const float* w_proj = (const float*)d_in[3];
  const float* b_proj = (const float*)d_in[4];
  float* out = (float*)d_out;

  char* ws = (char*)d_ws;
  const size_t QKV_BYTES = 3ull * 8 * 256 * 784 * 64 * 2;  // 616,562,688
  __bf16* qkv = (__bf16*)ws;
  __bf16* at  = (__bf16*)(ws + QKV_BYTES);

  qkv_win_kernel<<<dim3(12, 784, 8), 256, 0, stream>>>(x, w_qkv, b_qkv, qkv);
  win_attn_kernel<<<dim3(256, 8, 1), 256, 0, stream>>>(qkv, at);
  proj_kernel<<<dim3(4, 784, 8), 256, 0, stream>>>(at, w_proj, b_proj, out);
}
